// DistMatchLayer_v5_73461120631388
// MI455X (gfx1250) — matrix operand order repaired and checked by reading, not compiled, not run on hardware
//
#include <hip/hip_runtime.h>

namespace {

constexpr int   kNA   = 8192;
constexpr int   kNB   = 8192;
constexpr int   kC    = 256;
constexpr int   kNSeg = 4;
constexpr float kFullScale = 32.0f;

constexpr int kMTile = 128;  // A rows per block
constexpr int kNTile = 32;   // B columns staged per iteration
constexpr int kWaves = 8;    // 256 threads, wave32

typedef __attribute__((ext_vector_type(16))) __bf16 v16bf;
typedef __attribute__((ext_vector_type(8)))  __bf16 v8bf;
typedef __attribute__((ext_vector_type(8)))  float  v8f;
typedef __attribute__((ext_vector_type(4)))  float  v4f;

__device__ inline v8f v8f_zero() {
  v8f v;
#pragma unroll
  for (int i = 0; i < 8; ++i) v[i] = 0.0f;
  return v;
}

// Per CDNA5 ISA (CDNA5 ISA): 16-bit A-matrix 16x32 fragment — lanes 0-15 hold
// K=0..7 (VGPR0-3) and K=16..23 (VGPR4-7); lanes 16-31 hold K=8..15 / K=24..31.
// Here each lane loads K = hi*16..+15 contiguous: two 8-element (16B) chunks at +hi*16 and +hi*16+8.
__device__ inline v16bf load_afrag(const __bf16* rowp, int hi) {
  const v8bf lo = *(const v8bf*)(rowp + hi * 16);
  const v8bf up = *(const v8bf*)(rowp + hi * 16 + 8);
  v16bf r;
#pragma unroll
  for (int i = 0; i < 8; ++i) { r[i] = lo[i]; r[i + 8] = up[i]; }
  return r;
}

__global__ __launch_bounds__(256, 1)
void dist_match_fused(const float* __restrict__ feat_a,
                      const float* __restrict__ feat_b,
                      const int* __restrict__ coord_a,
                      const int* __restrict__ coord_b,
                      const long long* __restrict__ idx_a,
                      const long long* __restrict__ idx_b,
                      float* __restrict__ out) {
  __shared__ __bf16 sB [kNTile * kC];          // feat_b tile, [j][c]  (S gemm B)
  __shared__ __bf16 sBT[kC * kNTile];          // transposed,  [c][j]  (O gemm B)
  __shared__ float  sCB[kNTile * 4];           // per-col (x,y,z,|.|^2)
  __shared__ float  sCA[kWaves][16 * 4];       // per-row (x,y,z,|.|^2)
  __shared__ __bf16 sQ [kWaves][16 * kNTile];  // per-wave C-layout -> A-layout staging
  __shared__ int    sBnd[kNSeg + 1];

  const int tid  = threadIdx.x;
  const int lane = tid & 31;
  const int wave = tid >> 5;
  const int hi   = lane >> 4;   // which 16-lane half
  const int ln   = lane & 15;
  const int m0   = blockIdx.x * kMTile;
  const int row0 = m0 + wave * 16;

  // Segment boundaries of sorted idx_b (column ranges are contiguous).
  if (tid == 0) {
    sBnd[0] = 0;
    sBnd[kNSeg] = kNB;
#pragma unroll
    for (int s = 1; s < kNSeg; ++s) {
      int lo = 0, hh = kNB;
      while (lo < hh) {
        int mid = (lo + hh) >> 1;
        if (idx_b[mid] < (long long)s) lo = mid + 1; else hh = mid;
      }
      sBnd[s] = lo;
    }
  }

  // Row coordinates for this wave's 16 rows.
  if (lane < 16) {
    const int r = row0 + lane;
    const float x = (float)coord_a[r * 3 + 0];
    const float y = (float)coord_a[r * 3 + 1];
    const float z = (float)coord_a[r * 3 + 2];
    sCA[wave][lane * 4 + 0] = x;
    sCA[wave][lane * 4 + 1] = y;
    sCA[wave][lane * 4 + 2] = z;
    sCA[wave][lane * 4 + 3] = x * x + y * y + z * z;
  }

  // Segment id of the 8 rows this lane's C-layout slots hold (row = r + 8*hi).
  int segA[8];
#pragma unroll
  for (int r = 0; r < 8; ++r) segA[r] = (int)idx_a[row0 + r + 8 * hi];

  // Whole K=256 bf16 A strip in registers: 8 WMMA A fragments (16x32 each).
  v16bf aF[8];
#pragma unroll
  for (int kc = 0; kc < 8; ++kc) {
    const float* ap = feat_a + (size_t)(row0 + ln) * kC + kc * 32 + hi * 16;
    const v4f f0 = *(const v4f*)(ap + 0);
    const v4f f1 = *(const v4f*)(ap + 4);
    const v4f f2 = *(const v4f*)(ap + 8);
    const v4f f3 = *(const v4f*)(ap + 12);
#pragma unroll
    for (int e = 0; e < 4; ++e) {
      aF[kc][e]      = (__bf16)f0[e];
      aF[kc][e + 4]  = (__bf16)f1[e];
      aF[kc][e + 8]  = (__bf16)f2[e];
      aF[kc][e + 12] = (__bf16)f3[e];
    }
  }

  // Output accumulator: 16 rows x 256 cols f32 = 16 C-layout tiles.
  v8f O[16];
#pragma unroll
  for (int t = 0; t < 16; ++t) O[t] = v8f_zero();
  float mrow[8], lrow[8];
#pragma unroll
  for (int r = 0; r < 8; ++r) { mrow[r] = -1e30f; lrow[r] = 0.0f; }

  const int segLo = (int)idx_a[m0];
  const int segHi = (int)idx_a[m0 + kMTile - 1];
  __syncthreads();
  const int bnd1 = sBnd[1], bnd2 = sBnd[2], bnd3 = sBnd[3];
  const int jBeg = sBnd[segLo] & ~(kNTile - 1);
  int jEnd = (sBnd[segHi + 1] + kNTile - 1) & ~(kNTile - 1);
  if (jEnd > kNB) jEnd = kNB;

  for (int jt = jBeg; jt < jEnd; jt += kNTile) {
    // ---- cooperative stage: 32x256 feat_b tile fp32 -> bf16, both layouts ----
    {
      const int j  = tid >> 3;            // 0..31
      const int cc = (tid & 7) * 32;      // 0..224
      const float* bp = feat_b + (size_t)(jt + j) * kC + cc;
#pragma unroll
      for (int q8 = 0; q8 < 8; ++q8) {
        const v4f f = *(const v4f*)(bp + q8 * 4);
#pragma unroll
        for (int e = 0; e < 4; ++e) {
          const __bf16 b = (__bf16)f[e];
          const int c = cc + q8 * 4 + e;
          sB [j * kC + c] = b;
          sBT[c * kNTile + j] = b;
        }
      }
      if (jt + kNTile < jEnd)
        __builtin_prefetch(feat_b + (size_t)(jt + kNTile + j) * kC + cc, 0, 1);
      if (tid < kNTile) {
        const int jj = jt + tid;
        const float x = (float)coord_b[jj * 3 + 0];
        const float y = (float)coord_b[jj * 3 + 1];
        const float z = (float)coord_b[jj * 3 + 2];
        sCB[tid * 4 + 0] = x;
        sCB[tid * 4 + 1] = y;
        sCB[tid * 4 + 2] = z;
        sCB[tid * 4 + 3] = x * x + y * y + z * z;
      }
    }
    __syncthreads();

    // ---- S = Astrip @ Btile^T : two 16x16 f32 tiles, 16 WMMAs ----
    v8f S0 = v8f_zero();
    v8f S1 = v8f_zero();
#pragma unroll
    for (int kc = 0; kc < 8; ++kc) {
      // B fragment: lane holds col j = ln (+16 for tile1), K = hi*16..+15 contiguous.
      const v16bf b0 = *(const v16bf*)&sB[(ln)      * kC + kc * 32 + hi * 16];
      const v16bf b1 = *(const v16bf*)&sB[(16 + ln) * kC + kc * 32 + hi * 16];
      S0 = __builtin_amdgcn_wmma_f32_16x16x32_bf16(false, aF[kc], false, b0,
                                                   (short)0, S0, false, false);
      S1 = __builtin_amdgcn_wmma_f32_16x16x32_bf16(false, aF[kc], false, b1,
                                                   (short)0, S1, false, false);
    }

    // ---- masked online softmax + geometric weighting ----
    const int col0 = jt + ln;
    const int col1 = col0 + 16;
    const int sg0 = (col0 >= bnd1) + (col0 >= bnd2) + (col0 >= bnd3);
    const int sg1 = (col1 >= bnd1) + (col1 >= bnd2) + (col1 >= bnd3);
    const float cb0x = sCB[ln * 4 + 0], cb0y = sCB[ln * 4 + 1];
    const float cb0z = sCB[ln * 4 + 2], cb0n = sCB[ln * 4 + 3];
    const float cb1x = sCB[(16 + ln) * 4 + 0], cb1y = sCB[(16 + ln) * 4 + 1];
    const float cb1z = sCB[(16 + ln) * 4 + 2], cb1n = sCB[(16 + ln) * 4 + 3];
#pragma unroll
    for (int r = 0; r < 8; ++r) {
      float s0 = (segA[r] == sg0) ? S0[r] : -1e30f;
      float s1 = (segA[r] == sg1) ? S1[r] : -1e30f;
      // Row r+8*hi lives across the 16 lanes of this half; xor masks < 16 stay in-half.
      float rmax = fmaxf(s0, s1);
      rmax = fmaxf(rmax, __shfl_xor(rmax, 1, 32));
      rmax = fmaxf(rmax, __shfl_xor(rmax, 2, 32));
      rmax = fmaxf(rmax, __shfl_xor(rmax, 4, 32));
      rmax = fmaxf(rmax, __shfl_xor(rmax, 8, 32));
      const float mnew = fmaxf(mrow[r], rmax);
      const float scl  = __expf(mrow[r] - mnew);
      mrow[r] = mnew;
      const float p0 = __expf(s0 - mnew);
      const float p1 = __expf(s1 - mnew);
      float rs = p0 + p1;
      rs += __shfl_xor(rs, 1, 32);
      rs += __shfl_xor(rs, 2, 32);
      rs += __shfl_xor(rs, 4, 32);
      rs += __shfl_xor(rs, 8, 32);
      lrow[r] = lrow[r] * scl + rs;
#pragma unroll
      for (int t = 0; t < 16; ++t) O[t][r] *= scl;

      const int rr = r + 8 * hi;
      const float cax = sCA[wave][rr * 4 + 0], cay = sCA[wave][rr * 4 + 1];
      const float caz = sCA[wave][rr * 4 + 2], can = sCA[wave][rr * 4 + 3];
      const float d20 = can + cb0n - 2.0f * (cax * cb0x + cay * cb0y + caz * cb0z);
      const float d21 = can + cb1n - 2.0f * (cax * cb1x + cay * cb1y + caz * cb1z);
      const float dc0 = 1.0f - fminf(sqrtf(fmaxf(d20, 0.0f)) * (1.0f / kFullScale), 1.0f);
      const float dc1 = 1.0f - fminf(sqrtf(fmaxf(d21, 0.0f)) * (1.0f / kFullScale), 1.0f);
      sQ[wave][rr * kNTile + ln]      = (__bf16)(p0 * dc0);
      sQ[wave][rr * kNTile + 16 + ln] = (__bf16)(p1 * dc1);
    }
    // In-wave LDS store->load (cross-lane transpose) ordering.
    asm volatile("s_wait_dscnt 0" ::: "memory");

    // ---- O += Q(16x32) @ Btile(32x256): 16 WMMAs across the C dimension ----
    const v16bf qF = load_afrag(&sQ[wave][ln * kNTile], hi);
#pragma unroll
    for (int t = 0; t < 16; ++t) {
      // B fragment: lane holds col c = t*16+ln, K(j) = hi*16..+15 contiguous in sBT.
      const v16bf bT = *(const v16bf*)&sBT[(t * 16 + ln) * kNTile + hi * 16];
      O[t] = __builtin_amdgcn_wmma_f32_16x16x32_bf16(false, qF, false, bT,
                                                     (short)0, O[t], false, false);
    }
    __syncthreads();
  }

  // ---- epilogue: tmp = O / l, out = [feat_a | tmp] ----
#pragma unroll
  for (int r = 0; r < 8; ++r) {
    const float inv = (lrow[r] > 0.0f) ? (1.0f / lrow[r]) : 0.0f;
    float* op = out + (size_t)(row0 + r + 8 * hi) * (2 * kC) + kC;
#pragma unroll
    for (int t = 0; t < 16; ++t) op[t * 16 + ln] = O[t][r] * inv;
  }
  for (int i = tid; i < kMTile * kC; i += 256) {
    const int row = m0 + (i >> 8);
    const int c = i & (kC - 1);
    out[(size_t)row * (2 * kC) + c] = feat_a[(size_t)row * kC + c];
  }
}

}  // namespace

extern "C" void kernel_launch(void* const* d_in, const int* in_sizes, int n_in,
                              void* d_out, int out_size, void* d_ws, size_t ws_size,
                              hipStream_t stream) {
  (void)in_sizes; (void)n_in; (void)out_size; (void)d_ws; (void)ws_size;
  const float* feat_a = (const float*)d_in[0];
  const float* feat_b = (const float*)d_in[1];
  const int* coord_a  = (const int*)d_in[2];
  const int* coord_b  = (const int*)d_in[3];
  const long long* idx_a = (const long long*)d_in[4];
  const long long* idx_b = (const long long*)d_in[5];
  float* out = (float*)d_out;

  dim3 grid(kNA / kMTile);  // 64 blocks
  dim3 block(256);          // 8 waves (wave32)
  hipLaunchKernelGGL(dist_match_fused, grid, block, 0, stream,
                     feat_a, feat_b, coord_a, coord_b, idx_a, idx_b, out);
}
